// MNISTSum3Net_46677704573452
// MI455X (gfx1250) — matrix operand order repaired and checked by reading, not compiled, not run on hardware
//
#include <hip/hip_runtime.h>

// ---------------------------------------------------------------------------
// CDNA5 (gfx1250) wave32 WMMA types
// ---------------------------------------------------------------------------
typedef __attribute__((ext_vector_type(16))) _Float16 v16h;
typedef __attribute__((ext_vector_type(8)))  _Float16 v8h;
typedef __attribute__((ext_vector_type(8)))  float    v8f;

// A-matrix 16x32 f16 fragment (row-major M x K source).
// Lane L<16: row M=L, halves = K[k0+0..7] ++ K[k0+16..23]
// Lane L>=16: row M=L-16, halves = K[k0+8..15] ++ K[k0+24..31]
__device__ __forceinline__ v16h frag_a(const _Float16* row, int k0, int lane) {
  const _Float16* p = row + k0 + ((lane & 16) ? 8 : 0);
  v8h lo = *(const v8h*)p;
  v8h hi = *(const v8h*)(p + 16);
  v16h f;
#pragma unroll
  for (int i = 0; i < 8; ++i) { f[i] = lo[i]; f[i + 8] = hi[i]; }
  return f;
}

// B-matrix 32x16 f16 fragment, source stored as N x K rows (B^T row-major).
// Lane L<16: col N=L, halves = K[k0+0..7] ++ K[k0+16..23]; lane L>=16: N=L-16, K[k0+8..15] ++ K[k0+24..31]
__device__ __forceinline__ v16h frag_b(const _Float16* row, int k0, int lane) {
  const _Float16* p = row + k0 + ((lane & 16) ? 8 : 0);
  v8h lo = *(const v8h*)p;
  v8h hi = *(const v8h*)(p + 16);
  v16h f;
#pragma unroll
  for (int i = 0; i < 8; ++i) { f[i] = lo[i]; f[i + 8] = hi[i]; }
  return f;
}

__device__ __forceinline__ v8f wmma_f16(v16h a, v16h b, v8f c) {
  return __builtin_amdgcn_wmma_f32_16x16x32_f16(false, a, false, b, (short)0, c,
                                                false, false);
}

// ---------------------------------------------------------------------------
// Global -> LDS staging copy. Uses the gfx1250 async-to-LDS path (tracked by
// ASYNCcnt) when the builtin exists; falls back to b128 register copies.
// The builtin's exact parameter type (from hipcc diagnostic) is
//   'int __vector(4) __device__ *'  i.e. vector_size(16) int in AS1 / AS3.
// n_halves must be a multiple of 8; all threads must participate.
// ---------------------------------------------------------------------------
#if __has_builtin(__builtin_amdgcn_global_load_async_to_lds_b128)
typedef int v4i_vs __attribute__((vector_size(16)));
typedef __attribute__((address_space(1))) v4i_vs* g1_v4i_t;
typedef __attribute__((address_space(3))) v4i_vs* l3_v4i_t;
#endif

__device__ __forceinline__ void copy_g2l(const _Float16* __restrict__ g,
                                         _Float16* l, int n_halves,
                                         int tid, int nthreads) {
#if __has_builtin(__builtin_amdgcn_global_load_async_to_lds_b128)
  for (int e = tid * 8; e < n_halves; e += nthreads * 8) {
    __builtin_amdgcn_global_load_async_to_lds_b128(
        (g1_v4i_t)(unsigned long long)(g + e),
        (l3_v4i_t)(unsigned)(unsigned long long)(l + e), 0, 0);
  }
#if __has_builtin(__builtin_amdgcn_s_wait_asynccnt)
  __builtin_amdgcn_s_wait_asynccnt(0);
#else
  asm volatile("s_wait_asynccnt 0" ::: "memory");
#endif
#else
  for (int e = tid * 8; e < n_halves; e += nthreads * 8)
    *(v8h*)(l + e) = *(const v8h*)(g + e);
#endif
}

// ---------------------------------------------------------------------------
// f32 -> f16 weight conversion
// ---------------------------------------------------------------------------
__global__ void cvt_f32_to_f16_kernel(const float* __restrict__ src,
                                      _Float16* __restrict__ dst, int n) {
  int i = blockIdx.x * blockDim.x + threadIdx.x;
  if (i < n) dst[i] = (_Float16)src[i];
}

// ---------------------------------------------------------------------------
// conv1 (1->32ch, 5x5, valid) + bias + 2x2 maxpool; one block per image.
// Image (784 f32) and weights (800 f32) staged in LDS; f16 output.
// act1 layout: [net*B + img][c*144 + y*12 + x]   (12x12 pooled maps)
// ---------------------------------------------------------------------------
__global__ void __launch_bounds__(256)
conv1_pool_kernel(const float* __restrict__ a,
                  const float* __restrict__ bimg,
                  const float* __restrict__ cimg,
                  const float* __restrict__ w,
                  const float* __restrict__ bias,
                  _Float16* __restrict__ act1, int B) {
  __shared__ float img[784];
  __shared__ float wsm[800];
  __shared__ float bsm[32];
  int r = blockIdx.x;                      // 0 .. 3B-1
  int net = (r >= 2 * B) ? 2 : ((r >= B) ? 1 : 0);
  int im = r - net * B;
  const float* base = (net == 0) ? a : ((net == 1) ? bimg : cimg);
  const float* src = base + (long)im * 784;
  int tid = threadIdx.x;
  for (int e = tid; e < 784; e += 256) img[e] = src[e];
  for (int e = tid; e < 800; e += 256) wsm[e] = w[e];
  if (tid < 32) bsm[tid] = bias[tid];
  __syncthreads();
  for (int o = tid; o < 4608; o += 256) {
    int oc = o / 144;
    int rem = o - oc * 144;
    int py = rem / 12, px = rem - py * 12;
    const float* wk = wsm + oc * 25;
    float m = -1e30f;
#pragma unroll
    for (int dy = 0; dy < 2; ++dy) {
#pragma unroll
      for (int dx = 0; dx < 2; ++dx) {
        int oy = py * 2 + dy, ox = px * 2 + dx;
        float s = bsm[oc];
#pragma unroll
        for (int ky = 0; ky < 5; ++ky)
#pragma unroll
          for (int kx = 0; kx < 5; ++kx)
            s += img[(oy + ky) * 28 + (ox + kx)] * wk[ky * 5 + kx];
        m = fmaxf(m, s);
      }
    }
    act1[(long)r * 4608 + o] = (_Float16)m;
  }
}

// ---------------------------------------------------------------------------
// conv2 as implicit GEMM (M=64 out-ch, K=800=25*32, N=64 positions/image)
// One block = one image, 512 threads = 16 waves, wave -> (mt, nt) 16x16 tile.
// Whole conv1 image (9KB) staged in LDS once (async path if available);
// im2col stage is then LDS->LDS. WMMA f16->f32; 2x2 maxpool + bias.
// act2 layout: [imgrow][c*16 + y*4 + x]  (flattened 1024)
// ---------------------------------------------------------------------------
__global__ void __launch_bounds__(512)
conv2_wmma_kernel(const _Float16* __restrict__ act1,
                  const _Float16* __restrict__ w2h,
                  const float* __restrict__ b2,
                  _Float16* __restrict__ act2) {
  __shared__ _Float16 img[4608];        // conv1 activations for this image
  __shared__ _Float16 patch[64 * 32];   // [pos][k] im2col stage
  __shared__ float dbuf[16 * 256];      // per-wave 16x16 f32 D tile
  int imgrow = blockIdx.x;
  int tid = threadIdx.x;
  int wave = tid >> 5, lane = tid & 31;
  int mt = wave >> 2, nt = wave & 3;    // 4 m-tiles x 4 n-tiles
  const _Float16* src = act1 + (long)imgrow * 4608;
  copy_g2l(src, img, 4608, tid, 512);
  __syncthreads();
  const _Float16* arow = w2h + (long)(mt * 16 + (lane & 15)) * 800;
  const _Float16* brow = patch + (nt * 16 + (lane & 15)) * 32;
  v8f acc = {};
  for (int ks = 0; ks < 25; ++ks) {
    // im2col gather from LDS image: 2048 f16 elements, 4 per thread
    for (int e = tid; e < 2048; e += 512) {
      int pos = e >> 5, kk = e & 31;
      int K = ks * 32 + kk;
      int c = K / 25;
      int j = K - c * 25;
      int ky = j / 5, kx = j - ky * 5;
      int py = pos >> 3, px = pos & 7;
      patch[pos * 32 + kk] = img[c * 144 + (py + ky) * 12 + (px + kx)];
    }
    __syncthreads();
    v16h af = frag_a(arow, ks * 32, lane);
    v16h bf = frag_b(brow, 0, lane);
    acc = wmma_f16(af, bf, acc);
    __syncthreads();
  }
  // spill D tile to LDS for cross-lane 2x2 maxpool
  {
    int n = lane & 15;
    int mb = (lane & 16) ? 8 : 0;
#pragma unroll
    for (int r = 0; r < 8; ++r) dbuf[wave * 256 + (mb + r) * 16 + n] = acc[r];
  }
  __syncthreads();
  // tile holds channels mt*16..+15 at positions py in {2nt,2nt+1}, px 0..7
  for (int o = lane; o < 64; o += 32) {
    int m = o >> 2, x = o & 3;
    const float* t = dbuf + wave * 256 + m * 16;
    float v = fmaxf(fmaxf(t[2 * x], t[2 * x + 1]),
                    fmaxf(t[2 * x + 8], t[2 * x + 9]));
    int ch = mt * 16 + m;
    act2[(long)imgrow * 1024 + ch * 16 + nt * 4 + x] = (_Float16)(v + b2[ch]);
  }
}

// ---------------------------------------------------------------------------
// fc1: D[1024 x R] = fc1_w(f16)[1024x1024] * act2^T, + bias, ReLU -> act3 f16
// Block 256 = 8 waves; wave -> m-tile, block.y -> n-tile.
// B tile (16 rows x 1024, 32KB) staged once in LDS and shared by all 8 waves.
// ---------------------------------------------------------------------------
__global__ void __launch_bounds__(256)
fc1_wmma_kernel(const _Float16* __restrict__ act2,
                const _Float16* __restrict__ w1h,
                const float* __restrict__ b1,
                _Float16* __restrict__ act3) {
  __shared__ _Float16 btile[16 * 1024];  // 32KB shared B tile
  int tid = threadIdx.x;
  int wave = tid >> 5, lane = tid & 31;
  int mt = blockIdx.x * 8 + wave;        // 0..63 (1024 features)
  int ntile = blockIdx.y;                // 0..R/16-1
  copy_g2l(act2 + (long)ntile * 16 * 1024, btile, 16 * 1024, tid, 256);
  __syncthreads();
  const _Float16* arow = w1h + (long)(mt * 16 + (lane & 15)) * 1024;
  const _Float16* brow = btile + (lane & 15) * 1024;
  v8f acc = {};
#pragma unroll 4
  for (int k0 = 0; k0 < 1024; k0 += 32) {
    __builtin_prefetch(arow + k0 + 64, 0, 1);   // global_prefetch_b8
    v16h af = frag_a(arow, k0, lane);
    v16h bf = frag_b(brow, k0, lane);
    acc = wmma_f16(af, bf, acc);
  }
  int nl = lane & 15;
  int f0 = mt * 16 + ((lane & 16) ? 8 : 0);
  long row = (long)ntile * 16 + nl;
  v8h outv;
#pragma unroll
  for (int r = 0; r < 8; ++r) {
    float v = acc[r] + b1[f0 + r];
    outv[r] = (_Float16)fmaxf(v, 0.0f);
  }
  *(v8h*)(act3 + row * 1024 + f0) = outv;
}

// ---------------------------------------------------------------------------
// fc2 (10x1024) + softmax -> probs f32 [R][10]. One wave per row; each lane
// owns a contiguous 32-element chunk so loads vectorize to b128.
// ---------------------------------------------------------------------------
__global__ void __launch_bounds__(256)
fc2_softmax_kernel(const _Float16* __restrict__ act3,
                   const float* __restrict__ w2,
                   const float* __restrict__ b2,
                   float* __restrict__ probs) {
  int tid = threadIdx.x;
  int wave = tid >> 5, lane = tid & 31;
  long row = (long)blockIdx.x * 8 + wave;
  const _Float16* x = act3 + row * 1024 + lane * 32;
  float xk[32];
#pragma unroll
  for (int i = 0; i < 32; ++i) xk[i] = (float)x[i];
  float lg[10];
#pragma unroll
  for (int o = 0; o < 10; ++o) {
    const float* wr = w2 + o * 1024 + lane * 32;
    float s = 0.0f;
#pragma unroll
    for (int i = 0; i < 32; ++i) s += xk[i] * wr[i];
#pragma unroll
    for (int off = 16; off >= 1; off >>= 1) s += __shfl_xor(s, off, 32);
    lg[o] = s + b2[o];                  // same value on all lanes
  }
  if (lane == 0) {
    float m = -1e30f;
#pragma unroll
    for (int o = 0; o < 10; ++o) m = fmaxf(m, lg[o]);
    float e[10], sum = 0.0f;
#pragma unroll
    for (int o = 0; o < 10; ++o) { e[o] = __expf(lg[o] - m); sum += e[o]; }
    float inv = 1.0f / sum;
#pragma unroll
    for (int o = 0; o < 10; ++o) probs[row * 10 + o] = e[o] * inv;
  }
}

// ---------------------------------------------------------------------------
// Deterministic counter-hash PRNG + categorical sampling + histograms + MSE.
// One block per row b. Writes partial[b] = sum of 30 squared errors.
// ---------------------------------------------------------------------------
__device__ __forceinline__ unsigned long long mix64(unsigned long long x) {
  x ^= x >> 33; x *= 0xff51afd7ed558ccdULL;
  x ^= x >> 33; x *= 0xc4ceb9fe1a85ec53ULL;
  x ^= x >> 33;
  return x;
}
__device__ __forceinline__ float u01(unsigned long long x) {
  return (float)(unsigned)(mix64(x) >> 40) * (1.0f / 16777216.0f);
}
__device__ __forceinline__ int cat10(const float* cdf, float u) {
  int c = 9;
#pragma unroll
  for (int i = 9; i >= 0; --i) { if (u < cdf[i]) c = i; }
  return c;
}

__global__ void __launch_bounds__(256)
sample_loss_kernel(const float* __restrict__ probs, const int* __restrict__ y,
                   const int* __restrict__ n_ptr, float* __restrict__ partial,
                   int B) {
  __shared__ float dsh[30];
  __shared__ int hist[30];
  __shared__ float denomS;
  __shared__ float errs[30];
  int b = blockIdx.x, tid = threadIdx.x;
  if (tid < 30) {
    int net = tid / 10, cls = tid - net * 10;
    dsh[tid] = probs[((long)net * B + b) * 10 + cls];
    hist[tid] = 0;
  }
  __syncthreads();
  float cdf[30];
#pragma unroll
  for (int net = 0; net < 3; ++net) {
    float acc = 0.0f;
#pragma unroll
    for (int i = 0; i < 10; ++i) { acc += dsh[net * 10 + i]; cdf[net * 10 + i] = acc; }
  }
  int n = *n_ptr;
  int yb = y[b];
  for (int s = tid; s < n; s += blockDim.x) {
    unsigned long long base =
        ((unsigned long long)(unsigned)b << 21) | (unsigned long long)(unsigned)s;
    float ua = u01(base ^ 0x243F6A8885A308D3ULL);
    float ub = u01(base ^ 0x13198A2E03707344ULL);
    float uc = u01(base ^ 0xA4093822299F31D0ULL);
    int ca = cat10(cdf + 0, ua);
    int cb = cat10(cdf + 10, ub);
    int cc = cat10(cdf + 20, uc);
    if (ca + cb + cc == yb) {
      atomicAdd(&hist[ca], 1);
      atomicAdd(&hist[10 + cb], 1);
      atomicAdd(&hist[20 + cc], 1);
    }
  }
  __syncthreads();
  if (tid == 0) {
    int total = 0;
#pragma unroll
    for (int i = 0; i < 10; ++i) total += hist[i];
    denomS = fmaxf((float)total, 1.0f);
  }
  __syncthreads();
  if (tid < 30) {
    float p = (float)hist[tid] / denomS;
    float diff = dsh[tid] - p;
    errs[tid] = diff * diff;
  }
  __syncthreads();
  if (tid == 0) {
    float s = 0.0f;
#pragma unroll
    for (int i = 0; i < 30; ++i) s += errs[i];
    partial[b] = s;
  }
}

__global__ void __launch_bounds__(256)
reduce_mean_kernel(const float* __restrict__ partial, float* __restrict__ out,
                   int n, float scale) {
  __shared__ float sm[256];
  int tid = threadIdx.x;
  float s = 0.0f;
  for (int i = tid; i < n; i += 256) s += partial[i];
  sm[tid] = s;
  __syncthreads();
  for (int off = 128; off > 0; off >>= 1) {
    if (tid < off) sm[tid] += sm[tid + off];
    __syncthreads();
  }
  if (tid == 0) out[0] = sm[0] * scale;
}

// ---------------------------------------------------------------------------
// Host launch
// ---------------------------------------------------------------------------
extern "C" void kernel_launch(void* const* d_in, const int* in_sizes, int n_in,
                              void* d_out, int out_size, void* d_ws, size_t ws_size,
                              hipStream_t stream) {
  (void)n_in; (void)out_size; (void)ws_size;
  const float* a_imgs  = (const float*)d_in[0];
  const float* b_imgs  = (const float*)d_in[1];
  const float* c_imgs  = (const float*)d_in[2];
  const float* conv1_w = (const float*)d_in[3];
  const float* conv1_b = (const float*)d_in[4];
  const float* conv2_w = (const float*)d_in[5];
  const float* conv2_b = (const float*)d_in[6];
  const float* fc1_w   = (const float*)d_in[7];
  const float* fc1_b   = (const float*)d_in[8];
  const float* fc2_w   = (const float*)d_in[9];
  const float* fc2_b   = (const float*)d_in[10];
  const int*   y       = (const int*)d_in[11];
  const int*   n_samp  = (const int*)d_in[12];
  float* out = (float*)d_out;

  const int B = in_sizes[0] / 784;   // 2048
  const int R = 3 * B;               // 6144 rows total

  char* ws = (char*)d_ws;
  size_t off = 0;
  auto alloc = [&](size_t bytes) -> char* {
    char* p = ws + off;
    off = (off + bytes + 255) & ~(size_t)255;
    return p;
  };
  _Float16* w2h  = (_Float16*)alloc((size_t)64 * 800 * 2);     // conv2 weights f16
  _Float16* w1h  = (_Float16*)alloc((size_t)1024 * 1024 * 2);  // fc1 weights f16
  _Float16* act1 = (_Float16*)alloc((size_t)R * 4608 * 2);     // conv1 pooled
  _Float16* act2 = (_Float16*)alloc((size_t)R * 1024 * 2);     // conv2 pooled flat
  _Float16* act3 = (_Float16*)alloc((size_t)R * 1024 * 2);     // fc1 out
  float* probs   = (float*)alloc((size_t)R * 10 * 4);
  float* partial = (float*)alloc((size_t)B * 4);

  { int n = 64 * 800;
    cvt_f32_to_f16_kernel<<<(n + 255) / 256, 256, 0, stream>>>(conv2_w, w2h, n); }
  { int n = 1024 * 1024;
    cvt_f32_to_f16_kernel<<<(n + 255) / 256, 256, 0, stream>>>(fc1_w, w1h, n); }
  conv1_pool_kernel<<<R, 256, 0, stream>>>(a_imgs, b_imgs, c_imgs,
                                           conv1_w, conv1_b, act1, B);
  conv2_wmma_kernel<<<R, 512, 0, stream>>>(act1, w2h, conv2_b, act2);
  { dim3 g(8, (unsigned)(R / 16));
    fc1_wmma_kernel<<<g, 256, 0, stream>>>(act2, w1h, fc1_b, act3); }
  fc2_softmax_kernel<<<R / 8, 256, 0, stream>>>(act3, fc2_w, fc2_b, probs);
  sample_loss_kernel<<<B, 256, 0, stream>>>(probs, y, n_samp, partial, B);
  reduce_mean_kernel<<<1, 256, 0, stream>>>(partial, out, B,
                                            1.0f / (float)((long)R * 10));
}
